// TemporallyCalibratedAttention_42039139893865
// MI455X (gfx1250) — matrix operand order repaired and checked by reading, not compiled, not run on hardware
//
#include <hip/hip_runtime.h>

typedef __bf16 v16bf __attribute__((ext_vector_type(16)));
typedef float v8f __attribute__((ext_vector_type(8)));
typedef unsigned int v8u __attribute__((ext_vector_type(8)));

#define HD 12
#define DH 64
#define CDIM 768
#define NTOK 197
#define BATCH 64
#define NPAD 224
#define MTOT (BATCH * NTOK) /* 12608 = 197 * 64 row-tiles of 64 */

__device__ __forceinline__ unsigned short f2bf(float f) {
  unsigned int u = __builtin_bit_cast(unsigned int, f);
  u += 0x7FFFu + ((u >> 16) & 1u);  // round-to-nearest-even
  return (unsigned short)(u >> 16);
}

__device__ __forceinline__ v8f zero8() {
  v8f z;
#pragma unroll
  for (int i = 0; i < 8; ++i) z[i] = 0.0f;
  return z;
}

__device__ __forceinline__ v8f wmma_bf16(v16bf a, v16bf b, v8f c) {
  return __builtin_amdgcn_wmma_f32_16x16x32_bf16(false, a, false, b, (short)0, c,
                                                 false, false);
}

// LDS byte offset of a __shared__ object: low 32 bits of the generic pointer
// (LDS aperture maps addr[31:0] to the LDS byte address).
__device__ __forceinline__ unsigned lds_off(const void* p) {
  return (unsigned)(unsigned long long)p;
}

// CDNA5 async cache->LDS copy, tracked by ASYNCcnt (no VGPR round trip).
__device__ __forceinline__ void async_copy_b128(unsigned lds_byte_off,
                                                const void* gptr) {
  asm volatile("global_load_async_to_lds_b128 %0, %1, off" ::"v"(lds_byte_off),
               "v"(gptr)
               : "memory");
}
__device__ __forceinline__ void wait_async_le3() {
  asm volatile("s_wait_asynccnt 0x3" ::: "memory");
}
__device__ __forceinline__ void wait_async0() {
  asm volatile("s_wait_asynccnt 0x0" ::: "memory");
}

// A-matrix 16x32 bf16 fragment. ISA layout: lane m=L&15, h=L>>4;
// elems 0..7 -> K = h*8 + e, elems 8..15 -> K = 16 + h*8 + (e-8).
__device__ __forceinline__ v16bf load_frag_a(const unsigned short* base, int stride,
                                             int lane) {
  const int r = lane & 15, h = lane >> 4;
  const unsigned short* p = base + r * stride;
  v8u u;
#pragma unroll
  for (int i = 0; i < 8; ++i) {
    const int k = (i < 4) ? (h * 8 + 2 * i) : (16 + h * 8 + 2 * (i - 4));
    u[i] = *(const unsigned int*)(p + k);  // two contiguous bf16
  }
  return __builtin_bit_cast(v16bf, u);
}

// B-matrix 32x16 bf16 fragment, sourced from Bt (16 rows = columns of B).
// ISA layout: lane n=L&15 holds column n; elems 0..7 -> K = h*8 + e, elems 8..15 -> K = 16 + h*8 + (e-8).
__device__ __forceinline__ v16bf load_frag_b(const unsigned short* base, int stride,
                                             int lane) {
  const int n = lane & 15, h = lane >> 4;
  const unsigned short* p = base + n * stride;
  v8u u;
#pragma unroll
  for (int i = 0; i < 8; ++i) u[i] = *(const unsigned int*)(p + ((i < 4) ? (h * 8 + 2 * i) : (16 + h * 8 + 2 * (i - 4))));
  return __builtin_bit_cast(v16bf, u);
}

// Stage one 64x32 A-tile + 128x32 B-tile (bf16) into LDS buffers via async B128
// copies: 3 copies per thread (256 threads, 768 x 16B).
__device__ __forceinline__ void stage_async(const unsigned short* Ag,
                                            const unsigned short* Bg, int row0,
                                            int col0, int kc, unsigned asbase,
                                            unsigned bsbase, int tid) {
  {
    const int r = tid >> 2, seg = tid & 3;  // A: 64 rows x 4 x 16B
    async_copy_b128(asbase + r * 64 + seg * 16,
                    Ag + (size_t)(row0 + r) * CDIM + kc + seg * 8);
  }
#pragma unroll
  for (int q = 0; q < 2; ++q) {  // B: 128 rows x 4 x 16B
    const int e = tid + q * 256;
    const int r = e >> 2, seg = e & 3;
    async_copy_b128(bsbase + r * 64 + seg * 16,
                    Bg + (size_t)(col0 + r) * CDIM + kc + seg * 8);
  }
}

__global__ void cvt_kernel(const float* __restrict__ in,
                           unsigned short* __restrict__ out, int n) {
  for (int i = blockIdx.x * blockDim.x + threadIdx.x; i < n;
       i += gridDim.x * blockDim.x)
    out[i] = f2bf(in[i]);
}

// Temporal-context conv1d over CLS tokens: tc[bv*8+t][c] = conv(g)[bv][c][t]
__global__ void conv_tc_kernel(const float* __restrict__ x,
                               const float* __restrict__ wq,
                               const float* __restrict__ bq,
                               const float* __restrict__ wk,
                               const float* __restrict__ bk,
                               float* __restrict__ tcq, float* __restrict__ tck) {
  int idx = blockIdx.x * blockDim.x + threadIdx.x;
  if (idx >= 8 * 8 * CDIM) return;
  int c = idx % CDIM;
  int t = (idx / CDIM) & 7;
  int bv = idx / (CDIM * 8);
  float sq = bq[c], sk = bk[c];
  for (int j = 0; j < 3; ++j) {
    int tt = t - 1 + j;
    if (tt < 0 || tt >= 8) continue;
    const float* g = x + (size_t)(bv * 8 + tt) * NTOK * CDIM;  // CLS row
    const float* wqc = wq + (size_t)c * CDIM * 3 + j;
    const float* wkc = wk + (size_t)c * CDIM * 3 + j;
    for (int ci = 0; ci < CDIM; ++ci) {
      float v = g[ci];
      sq = fmaf(v, wqc[ci * 3], sq);
      sk = fmaf(v, wkc[ci * 3], sk);
    }
  }
  tcq[(bv * 8 + t) * CDIM + c] = sq;
  tck[(bv * 8 + t) * CDIM + c] = sk;
}

// qkv = x @ w_qkv^T + b_qkv; epilogue: q+tcq, k+tck -> bf16 (B,H,NPAD,DH),
// v -> transposed bf16 (B,H,DH,NPAD). 64x128 tile/block, 8 waves of 16x64,
// double-buffered async-to-LDS K pipeline.
__global__ __launch_bounds__(256) void qkv_gemm(
    const unsigned short* __restrict__ Axb, const unsigned short* __restrict__ Wb,
    const float* __restrict__ bqkv, const float* __restrict__ tcq,
    const float* __restrict__ tck, unsigned short* __restrict__ qb,
    unsigned short* __restrict__ kb, unsigned short* __restrict__ vt) {
  __shared__ __align__(16) unsigned short As[2][64 * 32];
  __shared__ __align__(16) unsigned short Bs[2][128 * 32];
  const int tid = threadIdx.x, lane = tid & 31, wave = tid >> 5;
  const int wr = wave & 3, wc = wave >> 2;
  const int row0 = blockIdx.x * 64, col0 = blockIdx.y * 128;
  const unsigned as0 = lds_off(&As[0][0]), bs0 = lds_off(&Bs[0][0]);
  v8f acc[4] = {zero8(), zero8(), zero8(), zero8()};
  stage_async(Axb, Wb, row0, col0, 0, as0, bs0, tid);
  for (int kc = 0; kc < CDIM; kc += 32) {
    const int cur = (kc >> 5) & 1;
    if (kc + 32 < CDIM) {  // prefetch next chunk into the other buffer
      stage_async(Axb, Wb, row0, col0, kc + 32, as0 + (cur ^ 1) * 4096u,
                  bs0 + (cur ^ 1) * 8192u, tid);
      wait_async_le3();  // in-order completion => current chunk landed
    } else {
      wait_async0();
    }
    __syncthreads();
    v16bf af = load_frag_a(&As[cur][0] + wr * 16 * 32, 32, lane);
#pragma unroll
    for (int ct = 0; ct < 4; ++ct) {
      v16bf bf = load_frag_b(&Bs[cur][0] + (wc * 64 + ct * 16) * 32, 32, lane);
      acc[ct] = wmma_bf16(af, bf, acc[ct]);
    }
    __syncthreads();  // all waves done with buf `cur` before it is overwritten
  }
  const int nloc = lane & 15;
#pragma unroll
  for (int ct = 0; ct < 4; ++ct) {
    const int j = col0 + wc * 64 + ct * 16 + nloc;  // 0..2303
    const int s = j / CDIM, cc = j % CDIM, hh = cc >> 6, d = cc & 63;
    const float bj = bqkv[j];
#pragma unroll
    for (int i = 0; i < 8; ++i) {
      const int mloc = (lane < 16) ? i : 8 + i;
      const int mg = row0 + wr * 16 + mloc;  // < 12608
      const int b = mg / NTOK, nn = mg - b * NTOK;
      float val = acc[ct][i] + bj;
      if (s == 0) {
        val += tcq[b * CDIM + cc];
        qb[((size_t)(b * HD + hh) * NPAD + nn) * DH + d] = f2bf(val);
      } else if (s == 1) {
        val += tck[b * CDIM + cc];
        kb[((size_t)(b * HD + hh) * NPAD + nn) * DH + d] = f2bf(val);
      } else {
        vt[((size_t)(b * HD + hh) * DH + d) * NPAD + nn] = f2bf(val);
      }
    }
  }
}

// Fused attention per (b,h): scores = qb.kb^T * 0.125, masked softmax, P.V.
// K/V^T async-staged into LDS (reused by all 8 waves); per-wave P buffer in LDS.
__global__ __launch_bounds__(256) void attn_kernel(
    const unsigned short* __restrict__ qb, const unsigned short* __restrict__ kb,
    const unsigned short* __restrict__ vt, unsigned short* __restrict__ ao) {
  __shared__ __align__(16) unsigned short ks[NPAD * DH];
  __shared__ __align__(16) unsigned short vs[DH * NPAD];
  __shared__ __align__(16) unsigned short ps[8 * 16 * NPAD];
  const int bh = blockIdx.x;
  const int b = bh / HD, hh = bh % HD;
  const unsigned short* qbase = qb + (size_t)bh * NPAD * DH;
  const unsigned short* kbase = kb + (size_t)bh * NPAD * DH;
  const unsigned short* vbase = vt + (size_t)bh * DH * NPAD;
  {
    const unsigned ko = lds_off(ks), vo = lds_off(vs);
    for (int e = threadIdx.x; e < NPAD * DH / 8; e += 256) {  // 1792 x 16B each
      async_copy_b128(ko + e * 16, (const char*)kbase + e * 16);
      async_copy_b128(vo + e * 16, (const char*)vbase + e * 16);
    }
    wait_async0();
  }
  __syncthreads();
  const int wave = threadIdx.x >> 5, lane = threadIdx.x & 31;
  const int nloc = lane & 15;
  unsigned short* psw = ps + wave * 16 * NPAD;
  for (int rt = wave; rt <= 12; rt += 8) {  // 13 query row-tiles cover N=197
    v16bf aq0 = load_frag_a(qbase + rt * 16 * DH, DH, lane);
    v16bf aq1 = load_frag_a(qbase + rt * 16 * DH + 32, DH, lane);
    v8f sc[14];
#pragma unroll
    for (int ct = 0; ct < 14; ++ct) {
      v8f s = zero8();
      s = wmma_bf16(aq0, load_frag_b(ks + ct * 16 * DH, DH, lane), s);
      s = wmma_bf16(aq1, load_frag_b(ks + ct * 16 * DH + 32, DH, lane), s);
      const float madd = (ct * 16 + nloc < NTOK) ? 0.0f : -1e30f;
#pragma unroll
      for (int i = 0; i < 8; ++i) sc[ct][i] = s[i] * 0.125f + madd;
    }
    float rmax[8], rinv[8];
#pragma unroll
    for (int i = 0; i < 8; ++i) {  // row max: 14 local + 16-lane butterfly
      float m = -1e30f;
#pragma unroll
      for (int ct = 0; ct < 14; ++ct) m = fmaxf(m, sc[ct][i]);
#pragma unroll
      for (int off = 8; off >= 1; off >>= 1) m = fmaxf(m, __shfl_xor(m, off, 16));
      rmax[i] = m;
    }
#pragma unroll
    for (int i = 0; i < 8; ++i) {
      float ssum = 0.0f;
#pragma unroll
      for (int ct = 0; ct < 14; ++ct) {
        float e = __expf(sc[ct][i] - rmax[i]);
        sc[ct][i] = e;
        ssum += e;
      }
#pragma unroll
      for (int off = 8; off >= 1; off >>= 1) ssum += __shfl_xor(ssum, off, 16);
      rinv[i] = 1.0f / ssum;
    }
#pragma unroll
    for (int ct = 0; ct < 14; ++ct) {  // dump P (bf16) to per-wave LDS, row-major
      const int col = ct * 16 + nloc;
#pragma unroll
      for (int i = 0; i < 8; ++i) {
        const int mrow = (lane < 16) ? i : 8 + i;
        psw[mrow * NPAD + col] = f2bf(sc[ct][i] * rinv[i]);
      }
    }
    asm volatile("s_wait_dscnt 0" ::: "memory");  // P visible to whole wave
#pragma unroll
    for (int dt = 0; dt < 4; ++dt) {  // out[16x64] = P(16x224) . V(224x64)
      v8f o = zero8();
#pragma unroll
      for (int kc = 0; kc < 7; ++kc) {
        v16bf ap = load_frag_a(psw + kc * 32, NPAD, lane);
        v16bf bv = load_frag_b(vs + dt * 16 * NPAD + kc * 32, NPAD, lane);
        o = wmma_bf16(ap, bv, o);
      }
      const int d = dt * 16 + nloc;
#pragma unroll
      for (int i = 0; i < 8; ++i) {
        const int mrow = (lane < 16) ? i : 8 + i;
        const int ng = rt * 16 + mrow;
        if (ng < NTOK)
          ao[(size_t)(b * NTOK + ng) * CDIM + hh * DH + d] = f2bf(o[i]);
      }
    }
  }
}

// out = ao @ w_proj^T + b_proj (fp32 output), same async double-buffered pipeline
__global__ __launch_bounds__(256) void proj_gemm(
    const unsigned short* __restrict__ A, const unsigned short* __restrict__ W,
    const float* __restrict__ bias, float* __restrict__ out) {
  __shared__ __align__(16) unsigned short As[2][64 * 32];
  __shared__ __align__(16) unsigned short Bs[2][128 * 32];
  const int tid = threadIdx.x, lane = tid & 31, wave = tid >> 5;
  const int wr = wave & 3, wc = wave >> 2;
  const int row0 = blockIdx.x * 64, col0 = blockIdx.y * 128;
  const unsigned as0 = lds_off(&As[0][0]), bs0 = lds_off(&Bs[0][0]);
  v8f acc[4] = {zero8(), zero8(), zero8(), zero8()};
  stage_async(A, W, row0, col0, 0, as0, bs0, tid);
  for (int kc = 0; kc < CDIM; kc += 32) {
    const int cur = (kc >> 5) & 1;
    if (kc + 32 < CDIM) {
      stage_async(A, W, row0, col0, kc + 32, as0 + (cur ^ 1) * 4096u,
                  bs0 + (cur ^ 1) * 8192u, tid);
      wait_async_le3();
    } else {
      wait_async0();
    }
    __syncthreads();
    v16bf af = load_frag_a(&As[cur][0] + wr * 16 * 32, 32, lane);
#pragma unroll
    for (int ct = 0; ct < 4; ++ct) {
      v16bf bf = load_frag_b(&Bs[cur][0] + (wc * 64 + ct * 16) * 32, 32, lane);
      acc[ct] = wmma_bf16(af, bf, acc[ct]);
    }
    __syncthreads();
  }
  const int nloc = lane & 15;
#pragma unroll
  for (int ct = 0; ct < 4; ++ct) {
    const int j = col0 + wc * 64 + ct * 16 + nloc;
    const float bj = bias[j];
#pragma unroll
    for (int i = 0; i < 8; ++i) {
      const int mloc = (lane < 16) ? i : 8 + i;
      const int mg = row0 + wr * 16 + mloc;
      out[(size_t)mg * CDIM + j] = acc[ct][i] + bj;
    }
  }
}

extern "C" void kernel_launch(void* const* d_in, const int* in_sizes, int n_in,
                              void* d_out, int out_size, void* d_ws, size_t ws_size,
                              hipStream_t stream) {
  const float* x = (const float*)d_in[0];
  const float* w_qkv = (const float*)d_in[1];
  const float* b_qkv = (const float*)d_in[2];
  const float* w_proj = (const float*)d_in[3];
  const float* b_proj = (const float*)d_in[4];
  const float* cqw = (const float*)d_in[5];
  const float* cqb = (const float*)d_in[6];
  const float* ckw = (const float*)d_in[7];
  const float* ckb = (const float*)d_in[8];
  (void)in_sizes; (void)n_in; (void)out_size; (void)ws_size;

  char* ws = (char*)d_ws;
  constexpr size_t SZ_XBF = (size_t)MTOT * CDIM * 2;
  constexpr size_t SZ_WQKV = (size_t)3 * CDIM * CDIM * 2;
  constexpr size_t SZ_WPROJ = (size_t)CDIM * CDIM * 2;
  constexpr size_t SZ_TC = (size_t)BATCH * CDIM * 4;
  constexpr size_t SZ_QB = (size_t)BATCH * HD * NPAD * DH * 2;
  constexpr size_t OFF_XBF = 0;
  constexpr size_t OFF_WQKV = OFF_XBF + SZ_XBF;
  constexpr size_t OFF_WPROJ = OFF_WQKV + SZ_WQKV;
  constexpr size_t OFF_TCQ = OFF_WPROJ + SZ_WPROJ;
  constexpr size_t OFF_TCK = OFF_TCQ + SZ_TC;
  constexpr size_t OFF_QB = OFF_TCK + SZ_TC;
  constexpr size_t OFF_KB = OFF_QB + SZ_QB;
  constexpr size_t OFF_VT = OFF_KB + SZ_QB;
  constexpr size_t OFF_AO = OFF_VT + SZ_QB;

  unsigned short* xbf = (unsigned short*)(ws + OFF_XBF);
  unsigned short* wqkvbf = (unsigned short*)(ws + OFF_WQKV);
  unsigned short* wprojbf = (unsigned short*)(ws + OFF_WPROJ);
  float* tcq = (float*)(ws + OFF_TCQ);
  float* tck = (float*)(ws + OFF_TCK);
  unsigned short* qbuf = (unsigned short*)(ws + OFF_QB);
  unsigned short* kbuf = (unsigned short*)(ws + OFF_KB);
  unsigned short* vbuf = (unsigned short*)(ws + OFF_VT);
  unsigned short* aobf = (unsigned short*)(ws + OFF_AO);

  // zero q/k/v pad rows (n = 197..223) — one contiguous async memset
  hipMemsetAsync(ws + OFF_QB, 0, SZ_QB * 3, stream);

  cvt_kernel<<<2048, 256, 0, stream>>>(x, xbf, MTOT * CDIM);
  cvt_kernel<<<256, 256, 0, stream>>>(w_qkv, wqkvbf, 3 * CDIM * CDIM);
  cvt_kernel<<<128, 256, 0, stream>>>(w_proj, wprojbf, CDIM * CDIM);
  conv_tc_kernel<<<192, 256, 0, stream>>>(x, cqw, cqb, ckw, ckb, tcq, tck);
  qkv_gemm<<<dim3(197, 18), 256, 0, stream>>>(xbf, wqkvbf, b_qkv, tcq, tck, qbuf,
                                              kbuf, vbuf);
  attn_kernel<<<BATCH * HD, 256, 0, stream>>>(qbuf, kbuf, vbuf, aobf);
  proj_gemm<<<dim3(197, 6), 256, 0, stream>>>(aobf, wprojbf, b_proj, (float*)d_out);
}
